// Decoder_33621003993892
// MI455X (gfx1250) — matrix operand order repaired and checked by reading, not compiled, not run on hardware
//
#include <hip/hip_runtime.h>

// ---------------------------------------------------------------------------
// Types for CDNA5 WMMA
// ---------------------------------------------------------------------------
typedef __attribute__((ext_vector_type(16))) __bf16 v16bf;
typedef __attribute__((ext_vector_type(8)))  float  v8f;

#define B_SZ   512
#define T_OUTC 100
#define T_INC  65
#define U_DIM  512
#define E_DIM  256
#define V_DIM  514
#define KIN    (E_DIM + U_DIM)   // 768 : concat(x, attn)
#define NTHREADS 512             // 16 wave32s -> 4 waves/SIMD for latency hiding
#define NWAVES   (NTHREADS / 32)

__device__ __forceinline__ __bf16 f2bf(float f) {
  union { float f; unsigned u; } x; x.f = f;
  unsigned r = (x.u + 0x7FFFu + ((x.u >> 16) & 1u)) >> 16;
  union { unsigned short s; __bf16 b; } y; y.s = (unsigned short)r;
  return y.b;
}
__device__ __forceinline__ float bf2f(__bf16 b) {
  union { __bf16 b; unsigned short s; } y; y.b = b;
  union { unsigned u; float f; } x; x.u = ((unsigned)y.s) << 16;
  return x.f;
}

// A-matrix (16x32 bf16) lane layout: row M = lane%16; lanes 0-15 hold
// K 0..15, lanes 16-31 hold K 16..31 (matches pack_b B layout).
__device__ __forceinline__ v16bf load_a_lds(const __bf16* base, int rowStride,
                                            int kBase, int lane) {
  const __bf16* p = base + (lane & 15) * rowStride + kBase + ((lane & 16) ? 16 : 0);
  union { v16bf v; __bf16 e[16]; } u;
#pragma unroll
  for (int i = 0; i < 8; ++i) { u.e[i] = p[i]; u.e[i + 8] = p[i + 8]; }
  return u.v;
}

// B tiles pre-packed so each lane reads its 16 bf16 (32B) contiguously.
__device__ __forceinline__ v16bf load_b_tile(const __bf16* w, int kt, int nt,
                                             int ntiles, int lane) {
  return *(const v16bf*)(w + (((size_t)kt * ntiles + nt) * 32 + lane) * 16);
}

__device__ __forceinline__ v8f wmma_bf16(v16bf a, v16bf b, v8f c) {
  return __builtin_amdgcn_wmma_f32_16x16x32_bf16(false, a, false, b, (short)0, c,
                                                 false, false);
}

// ---------------------------------------------------------------------------
// Pack a row-major [K][N] fp32 weight into bf16 WMMA B-tile layout.
// Lane l<16: K = kt*32 + e, N = nt*16 + l; lane l>=16: K = kt*32+16+e.
// ---------------------------------------------------------------------------
__global__ __launch_bounds__(256) void pack_b(const float* __restrict__ W,
                                              __bf16* __restrict__ dst,
                                              int K, int N, int Ntiles) {
  size_t idx = (size_t)blockIdx.x * 256 + threadIdx.x;
  size_t total = (size_t)(K / 32) * Ntiles * 32 * 16;
  if (idx >= total) return;
  int e    = (int)(idx & 15);
  int lane = (int)((idx >> 4) & 31);
  size_t tile = idx >> 9;
  int nt = (int)(tile % Ntiles);
  int kt = (int)(tile / Ntiles);
  int n  = nt * 16 + (lane & 15);
  int k  = kt * 32 + ((lane & 16) ? 16 : 0) + e;
  float v = (n < N) ? W[(size_t)k * N + n] : 0.f;
  dst[idx] = f2bf(v);
}

// ---------------------------------------------------------------------------
// keys = memory @ W_mem  : [B*T_IN, U] x [U, U] -> fp32 workspace
// One block = one 16-row M tile; 16 waves split the 32 N-tiles.
// ---------------------------------------------------------------------------
__global__ __launch_bounds__(NTHREADS) void keys_kernel(
    const float* __restrict__ memory, const __bf16* __restrict__ WmemP,
    float* __restrict__ keys) {
  __shared__ __bf16 s_a[16 * U_DIM];
  const int tid = threadIdx.x, lane = tid & 31, w = tid >> 5;
  const size_t row0 = (size_t)blockIdx.x * 16;
  for (int j = tid; j < 16 * U_DIM; j += NTHREADS)
    s_a[j] = f2bf(memory[(row0 + (j >> 9)) * U_DIM + (j & 511)]);
  __syncthreads();
  for (int nt = w; nt < 32; nt += NWAVES) {
    v8f acc = {};
#pragma unroll
    for (int kt = 0; kt < 16; ++kt)
      acc = wmma_bf16(load_a_lds(s_a, U_DIM, kt * 32, lane),
                      load_b_tile(WmemP, kt, nt, 32, lane), acc);
    int n  = nt * 16 + (lane & 15);
    int mb = (lane & 16) ? 8 : 0;
#pragma unroll
    for (int r = 0; r < 8; ++r)
      keys[(row0 + r + mb) * U_DIM + n] = acc[r];
  }
}

// ---------------------------------------------------------------------------
// Persistent decoder: one block owns 16 batch rows for all 100 steps.
// ---------------------------------------------------------------------------
__global__ __launch_bounds__(NTHREADS) void decoder_kernel(
    const int* __restrict__ inputs, const float* __restrict__ memory,
    const float* __restrict__ enc_h, const float* __restrict__ enc_c,
    const float* __restrict__ embed, const float* __restrict__ lstm_bias,
    const float* __restrict__ fc_b,
    const __bf16* __restrict__ WkP, const __bf16* __restrict__ WrP,
    const __bf16* __restrict__ WattnP, const __bf16* __restrict__ fcWP,
    const float* __restrict__ keys, float* __restrict__ out) {
  __shared__ __bf16 s_inp[16 * KIN];          // [x_t | attn]   bf16
  __shared__ __bf16 s_h[2][16 * U_DIM];       // double-buffered h (bf16)
  __shared__ __bf16 s_ctx[16 * U_DIM];        // context (bf16)
  __shared__ float  s_c[16 * U_DIM];          // cell state fp32
  __shared__ float  s_sc[16 * T_INC];         // scores / softmax

  const int tid  = threadIdx.x;
  const int lane = tid & 31;
  const int w    = tid >> 5;                  // wave 0..15
  const int b0   = blockIdx.x * 16;

  for (int j = tid; j < 16 * U_DIM; j += NTHREADS) {
    int r = j >> 9, u = j & 511;
    s_h[0][j] = f2bf(enc_h[(size_t)(b0 + r) * U_DIM + u]);
    s_c[j]    = enc_c[(size_t)(b0 + r) * U_DIM + u];
    s_inp[r * KIN + E_DIM + u] = f2bf(0.f);   // attn0 = 0
  }
  __syncthreads();

  for (int t = 0; t < T_OUTC; ++t) {
    const int p = t & 1;
    const __bf16* prv = s_h[p];
    __bf16* cur = s_h[p ^ 1];

    // ---- P0: embedding gather -> inp[:,0:256]
    for (int j = tid; j < 16 * E_DIM; j += NTHREADS) {
      int r = j >> 8, e = j & 255;
      int tok = inputs[(size_t)(b0 + r) * T_OUTC + t];
      s_inp[r * KIN + e] = f2bf(embed[(size_t)tok * E_DIM + e]);
    }
    __syncthreads();

    // ---- P1: z = inp@Wk + h@Wr + b ; fused LSTM gating (relu cell act)
    // Branch-free, fully unrolled K loops -> constant-offset global_load
    // clauses; hardware LOADcnt gives one-deep load/WMMA overlap.
    for (int ut = w; ut < 32; ut += NWAVES) {
      v8f ai = {}, af = {}, ag = {}, ao = {};
#pragma unroll
      for (int kt = 0; kt < 24; ++kt) {                 // K over concat(x,attn)
        v16bf a = load_a_lds(s_inp, KIN, kt * 32, lane);
        ai = wmma_bf16(a, load_b_tile(WkP, kt, ut,      128, lane), ai);
        af = wmma_bf16(a, load_b_tile(WkP, kt, ut + 32, 128, lane), af);
        ag = wmma_bf16(a, load_b_tile(WkP, kt, ut + 64, 128, lane), ag);
        ao = wmma_bf16(a, load_b_tile(WkP, kt, ut + 96, 128, lane), ao);
      }
#pragma unroll
      for (int kt = 0; kt < 16; ++kt) {                 // K over h_{t-1}
        v16bf a = load_a_lds(prv, U_DIM, kt * 32, lane);
        ai = wmma_bf16(a, load_b_tile(WrP, kt, ut,      128, lane), ai);
        af = wmma_bf16(a, load_b_tile(WrP, kt, ut + 32, 128, lane), af);
        ag = wmma_bf16(a, load_b_tile(WrP, kt, ut + 64, 128, lane), ag);
        ao = wmma_bf16(a, load_b_tile(WrP, kt, ut + 96, 128, lane), ao);
      }
      int n  = ut * 16 + (lane & 15);
      int mb = (lane & 16) ? 8 : 0;
      float bi = lstm_bias[n],        bfv = lstm_bias[512 + n];
      float bg = lstm_bias[1024 + n], bo  = lstm_bias[1536 + n];
#pragma unroll
      for (int r = 0; r < 8; ++r) {
        int m = r + mb;
        float zi = ai[r] + bi, zf = af[r] + bfv, zg = ag[r] + bg, zo = ao[r] + bo;
        float iv = 1.f / (1.f + __expf(-zi));
        float fv = 1.f / (1.f + __expf(-zf));
        float ov = 1.f / (1.f + __expf(-zo));
        float gv = fmaxf(zg, 0.f);
        float cn = fv * s_c[m * U_DIM + n] + iv * gv;
        float hn = ov * fmaxf(cn, 0.f);
        s_c[m * U_DIM + n] = cn;
        cur[m * U_DIM + n] = f2bf(hn);
      }
    }
    __syncthreads();

    // ---- P2a: score[b,t'] = h_new . keys[b,t']   (per-row matvec, VALU)
    for (int j = tid; j < 16 * T_INC; j += NTHREADS) {
      int r = j / T_INC, tt = j % T_INC;
      const float*  kp = keys + ((size_t)(b0 + r) * T_INC + tt) * U_DIM;
      const __bf16* hp = cur + r * U_DIM;
      float s = 0.f;
      for (int u = 0; u < U_DIM; u += 4) {
        float4 k4 = *(const float4*)(kp + u);
        s += bf2f(hp[u    ]) * k4.x;
        s += bf2f(hp[u + 1]) * k4.y;
        s += bf2f(hp[u + 2]) * k4.z;
        s += bf2f(hp[u + 3]) * k4.w;
      }
      s_sc[j] = s;
    }
    __syncthreads();

    // ---- P2b: softmax over T_IN (one thread per row)
    if (tid < 16) {
      float mx = -1e30f;
      for (int tt = 0; tt < T_INC; ++tt) mx = fmaxf(mx, s_sc[tid * T_INC + tt]);
      float sum = 0.f;
      for (int tt = 0; tt < T_INC; ++tt) {
        float e = __expf(s_sc[tid * T_INC + tt] - mx);
        s_sc[tid * T_INC + tt] = e;
        sum += e;
      }
      float inv = 1.f / sum;
      for (int tt = 0; tt < T_INC; ++tt) s_sc[tid * T_INC + tt] *= inv;
    }
    __syncthreads();

    // ---- P2c: ctx[b,u] = sum_t align * memory[b,t,u]
    for (int j = tid; j < 16 * U_DIM; j += NTHREADS) {
      int r = j >> 9, u = j & 511;
      const float* mp = memory + (size_t)(b0 + r) * T_INC * U_DIM + u;
      float s = 0.f;
      for (int tt = 0; tt < T_INC; ++tt)
        s += s_sc[r * T_INC + tt] * mp[(size_t)tt * U_DIM];
      s_ctx[j] = f2bf(s);
    }
    __syncthreads();

    // ---- P3: attn = concat(h_new, ctx) @ W_attn  -> inp[:,256:768]
    for (int nt = w; nt < 32; nt += NWAVES) {
      v8f acc = {};
#pragma unroll
      for (int kt = 0; kt < 16; ++kt)
        acc = wmma_bf16(load_a_lds(cur, U_DIM, kt * 32, lane),
                        load_b_tile(WattnP, kt, nt, 32, lane), acc);
#pragma unroll
      for (int kt = 0; kt < 16; ++kt)
        acc = wmma_bf16(load_a_lds(s_ctx, U_DIM, kt * 32, lane),
                        load_b_tile(WattnP, kt + 16, nt, 32, lane), acc);
      int n  = nt * 16 + (lane & 15);
      int mb = (lane & 16) ? 8 : 0;
#pragma unroll
      for (int r = 0; r < 8; ++r)
        s_inp[(r + mb) * KIN + E_DIM + n] = f2bf(acc[r]);
    }
    __syncthreads();

    // ---- P4: out = attn @ fc_W + fc_b   (N padded 514 -> 528)
    for (int nt = w; nt < 33; nt += NWAVES) {
      v8f acc = {};
#pragma unroll
      for (int kt = 0; kt < 16; ++kt)
        acc = wmma_bf16(load_a_lds(s_inp + E_DIM, KIN, kt * 32, lane),
                        load_b_tile(fcWP, kt, nt, 33, lane), acc);
      int n = nt * 16 + (lane & 15);
      if (n < V_DIM) {
        float bb = fc_b[n];
        int mb = (lane & 16) ? 8 : 0;
#pragma unroll
        for (int r = 0; r < 8; ++r)
          out[((size_t)(b0 + r + mb) * T_OUTC + t) * V_DIM + n] = acc[r] + bb;
      }
    }
    __syncthreads();
  }
}

// ---------------------------------------------------------------------------
extern "C" void kernel_launch(void* const* d_in, const int* in_sizes, int n_in,
                              void* d_out, int out_size, void* d_ws, size_t ws_size,
                              hipStream_t stream) {
  (void)in_sizes; (void)n_in; (void)out_size; (void)ws_size;
  const int*   inputs = (const int*)  d_in[0];
  const float* memory = (const float*)d_in[1];
  const float* enc_h  = (const float*)d_in[2];
  const float* enc_c  = (const float*)d_in[3];
  const float* embed  = (const float*)d_in[4];
  const float* W_mem  = (const float*)d_in[5];
  const float* W_attn = (const float*)d_in[6];
  const float* lstm_k = (const float*)d_in[7];
  const float* lstm_r = (const float*)d_in[8];
  const float* lstm_b = (const float*)d_in[9];
  const float* fc_W   = (const float*)d_in[10];
  const float* fc_b   = (const float*)d_in[11];
  float* out = (float*)d_out;

  char* ws = (char*)d_ws;
  size_t off = 0;
  auto take = [&](size_t bytes) {
    char* pp = ws + off;
    off = (off + bytes + 255) & ~(size_t)255;
    return pp;
  };
  __bf16* WkP   = (__bf16*)take((size_t)24 * 128 * 512 * 2);  // 768x2048
  __bf16* WrP   = (__bf16*)take((size_t)16 * 128 * 512 * 2);  // 512x2048
  __bf16* WattnP= (__bf16*)take((size_t)32 * 32  * 512 * 2);  // 1024x512
  __bf16* fcWP  = (__bf16*)take((size_t)16 * 33  * 512 * 2);  // 512x528(pad)
  __bf16* WmemP = (__bf16*)take((size_t)16 * 32  * 512 * 2);  // 512x512
  float*  keys  = (float*) take((size_t)B_SZ * T_INC * U_DIM * 4);

  // Pack weights into WMMA B-tile bf16 layout.
  pack_b<<<dim3((24u * 128 * 512 + 255) / 256), 256, 0, stream>>>(lstm_k, WkP,  768, 2048, 128);
  pack_b<<<dim3((16u * 128 * 512 + 255) / 256), 256, 0, stream>>>(lstm_r, WrP,  512, 2048, 128);
  pack_b<<<dim3((32u * 32  * 512 + 255) / 256), 256, 0, stream>>>(W_attn, WattnP, 1024, 512, 32);
  pack_b<<<dim3((16u * 33  * 512 + 255) / 256), 256, 0, stream>>>(fc_W,   fcWP, 512, 514, 33);
  pack_b<<<dim3((16u * 32  * 512 + 255) / 256), 256, 0, stream>>>(W_mem,  WmemP, 512, 512, 32);

  // keys = memory @ W_mem : (512*65)/16 = 2080 row tiles
  keys_kernel<<<dim3(2080), NTHREADS, 0, stream>>>(memory, WmemP, keys);

  // Persistent decoder: 512/16 = 32 blocks x 16 wave32s.
  decoder_kernel<<<dim3(B_SZ / 16), NTHREADS, 0, stream>>>(
      inputs, memory, enc_h, enc_c, embed, lstm_b, fc_b,
      WkP, WrP, WattnP, fcWP, keys, out);
}
